// GCN_LSTM_11562051960909
// MI455X (gfx1250) — matrix operand order repaired and checked by reading, not compiled, not run on hardware
//
#include <hip/hip_runtime.h>
#include <hip/hip_bf16.h>
#include <stdint.h>
#include <stddef.h>

// ---------------- Problem constants (from reference) ----------------
#define NN      23040      // N_NODES
#define EE      368640     // N_EDGES
#define NG      64         // graphs
#define TSTEPS  360        // nodes per graph
#define INDIM   3000
#define HID     256
#define GOUT    16
#define LH      16

#define CDIV(a,b) (((a)+(b)-1)/(b))

typedef __attribute__((ext_vector_type(16))) __bf16          v16bf;
typedef __attribute__((ext_vector_type(16))) unsigned short  v16u;
typedef __attribute__((ext_vector_type(8)))  float           v8f;

__device__ __forceinline__ unsigned short f2bf(float f) {
  // round-to-nearest (half away from zero on ties): 1 VALU op
  unsigned int u = __float_as_uint(f);
  return (unsigned short)((u + 0x8000u) >> 16);
}

// ---------------- utility kernels ----------------
__global__ void zero_f32(float* __restrict__ p, size_t n) {
  size_t i = (size_t)blockIdx.x * blockDim.x + threadIdx.x;
  if (i < n) p[i] = 0.0f;
}

__global__ void deg_accum(const int* __restrict__ dst, float* __restrict__ deg) {
  size_t i = (size_t)blockIdx.x * blockDim.x + threadIdx.x;
  if (i < (size_t)EE) atomicAdd(&deg[dst[i]], 1.0f);
}

__global__ void deg_finalize(const float* __restrict__ deg,
                             float* __restrict__ invs, float* __restrict__ invd) {
  size_t i = (size_t)blockIdx.x * blockDim.x + threadIdx.x;
  if (i < (size_t)NN) {
    float d = deg[i] + 1.0f;           // self loop
    invs[i] = rsqrtf(d);
    invd[i] = 1.0f / d;
  }
}

// Pack W[K][N] (fp32, row-major) into per-lane bf16 WMMA B fragments.
// P[((kt*ntiles + nt)*32 + lane)*16 + e], k = kt*32 + 16*(lane>>4) + e,
// n = nt*16 + (lane&15); zero-pad k >= K.
__global__ void pack_w_bf16(const float* __restrict__ W, unsigned short* __restrict__ P,
                            int K, int N, int ktiles, int ntiles) {
  size_t tid = (size_t)blockIdx.x * blockDim.x + threadIdx.x;
  size_t total = (size_t)ktiles * ntiles * 512;
  if (tid >= total) return;
  int e    = (int)(tid & 15);
  int lane = (int)((tid >> 4) & 31);
  size_t rest = tid >> 9;
  int nt = (int)(rest % ntiles);
  int kt = (int)(rest / ntiles);
  int k = kt * 32 + ((lane >> 4) << 4) + e;
  int n = nt * 16 + (lane & 15);
  float v = (k < K) ? W[(size_t)k * N + n] : 0.0f;
  P[tid] = f2bf(v);
}

// ---------------- WMMA GEMM: C[M=23040, N] = A[M,K] * W[K,N] ----------------
// Block = 256 threads = 8 waves = TM M-tiles x WN wave-columns; each wave owns
// NQ adjacent 16x16 N-tiles (NQ accumulators), reusing one A fragment for NQ
// WMMAs. TM*WN == 8. A panel double-buffered in LDS (stride 36 floats:
// 16B-aligned rows, conflict-free across the 16 rows read per lane group);
// one barrier per k-step, next tile fetched under the WMMA chain.
template<int TM, int WN, int NQ>
__global__ __launch_bounds__(256)
void gemm_wmma_bf16(const float* __restrict__ A, const unsigned short* __restrict__ Bp,
                    float* __restrict__ C, int K, int N, int ktiles, int ntiles) {
  __shared__ float ldsA[2][TM * 16 * 36];
  const int lane = threadIdx.x & 31;
  const int wv   = threadIdx.x >> 5;
  const int mt   = wv / WN;
  const int nw   = wv % WN;
  const int row  = lane & 15;
  const int grp  = lane >> 4;
  const int gM0  = blockIdx.x * (TM * 16);
  const int gNt0 = blockIdx.y * (WN * NQ) + nw * NQ;   // first N-tile of this wave

  // staging indices (fixed per thread): float4 granularity, K % 4 == 0
  int sr[TM / 2], skq[TM / 2];
#pragma unroll
  for (int i = 0; i < TM / 2; ++i) {
    int idx = threadIdx.x + i * 256;
    sr[i]  = idx >> 3;                   // row within panel
    skq[i] = idx & 7;                    // float4 within 32-k chunk
  }

  v8f acc[NQ];
#pragma unroll
  for (int q = 0; q < NQ; ++q) acc[q] = (v8f){};

  // incremental B pointer: fragment q at bp + q*512 ushorts (1 KB stride)
  const unsigned short* bp = Bp + ((size_t)gNt0 * 32 + lane) * 16;
  const size_t bstep = (size_t)ntiles * 512;

  // prologue: stage tile 0
#pragma unroll
  for (int i = 0; i < TM / 2; ++i) {
    int k = skq[i] * 4;
    float4 v = (k < K) ? *(const float4*)(A + (size_t)(gM0 + sr[i]) * K + k)
                       : make_float4(0.f, 0.f, 0.f, 0.f);
    *(float4*)&ldsA[0][sr[i] * 36 + skq[i] * 4] = v;
  }
  __syncthreads();

  for (int kt = 0; kt < ktiles; ++kt) {
    const int cur = kt & 1;

    // A fragment from current LDS buffer (contiguous k = 16*grp + e, matching packed B)
    union { v16u u; v16bf b; } af;
    const float* ar = &ldsA[cur][(mt * 16 + row) * 36];
#pragma unroll
    for (int e = 0; e < 16; ++e) {
      int kl = (grp << 4) + e;
      af.u[e] = f2bf(ar[kl]);
    }

    // all B fragments first (pipelined loads, constant offsets)
    v16u bu[NQ];
#pragma unroll
    for (int q = 0; q < NQ; ++q) bu[q] = *(const v16u*)(bp + q * 512);

    // issue next A tile's global loads under the WMMA chain
    float4 stg[TM / 2];
    const bool more = (kt + 1 < ktiles);
    if (more) {
#pragma unroll
      for (int i = 0; i < TM / 2; ++i) {
        int k = (kt + 1) * 32 + skq[i] * 4;
        stg[i] = (k < K) ? *(const float4*)(A + (size_t)(gM0 + sr[i]) * K + k)
                         : make_float4(0.f, 0.f, 0.f, 0.f);
      }
    }

    // WMMA chain: one A fragment x NQ B fragments
#pragma unroll
    for (int q = 0; q < NQ; ++q) {
      union { v16u u; v16bf b; } bf;
      bf.u = bu[q];
      acc[q] = __builtin_amdgcn_wmma_f32_16x16x32_bf16(false, af.b, false, bf.b,
                                                       (short)0, acc[q], false, false);
    }

    // store staged tile into alternate buffer
    if (more) {
#pragma unroll
      for (int i = 0; i < TM / 2; ++i)
        *(float4*)&ldsA[cur ^ 1][sr[i] * 36 + skq[i] * 4] = stg[i];
    }
    bp += bstep;
    __syncthreads();
  }

  // C layout: lanes 0-15 -> M=v, lanes 16-31 -> M=8+v; N = lane&15
  const int gm = gM0 + mt * 16 + 8 * grp;
#pragma unroll
  for (int q = 0; q < NQ; ++q) {
    const int gn = (gNt0 + q) * 16 + row;
#pragma unroll
    for (int v = 0; v < 8; ++v)
      C[(size_t)(gm + v) * N + gn] = acc[q][v];
  }
}

// ---------------- edge scatter: agg[dst] += h[src] * invs[src]*invs[dst] -----
template<int DIM>
__global__ void edge_scatter(const int* __restrict__ src, const int* __restrict__ dst,
                             const float* __restrict__ invs,
                             const float* __restrict__ H, float* __restrict__ AGG) {
  const int TPE = DIM / 4;
  size_t tid = (size_t)blockIdx.x * blockDim.x + threadIdx.x;
  size_t e = tid / TPE;
  if (e >= (size_t)EE) return;
  int f0 = (int)(tid % TPE) * 4;
  int s = src[e], d = dst[e];
  float w = invs[s] * invs[d];
  const float4 hv = *(const float4*)(H + (size_t)s * DIM + f0);
  float* a = AGG + (size_t)d * DIM + f0;
  atomicAdd(a + 0, hv.x * w);
  atomicAdd(a + 1, hv.y * w);
  atomicAdd(a + 2, hv.z * w);
  atomicAdd(a + 3, hv.w * w);
}

// out = relu(agg + h*inv_deg[row] + b[col])
template<int DIM>
__global__ void combine_relu(const float* __restrict__ AGG, const float* __restrict__ H,
                             const float* __restrict__ invd, const float* __restrict__ b,
                             float* __restrict__ OUT) {
  size_t tid = (size_t)blockIdx.x * blockDim.x + threadIdx.x;
  if (tid >= (size_t)NN * DIM) return;
  int row = (int)(tid / DIM);
  int col = (int)(tid % DIM);
  float v = AGG[tid] + H[tid] * invd[row] + b[col];
  OUT[tid] = v > 0.0f ? v : 0.0f;
}

// ---------------- bidirectional LSTM (one block per graph x direction) -------
__global__ __launch_bounds__(64)
void lstm_kernel(const float* __restrict__ seq,
                 const float* __restrict__ Wih_f, const float* __restrict__ Whh_f,
                 const float* __restrict__ bih_f, const float* __restrict__ bhh_f,
                 const float* __restrict__ Wih_b, const float* __restrict__ Whh_b,
                 const float* __restrict__ bih_b, const float* __restrict__ bhh_b,
                 float* __restrict__ hn) {
  const int g   = blockIdx.x;     // graph
  const int dir = blockIdx.y;     // 0=fwd, 1=bwd
  const int r   = threadIdx.x;    // gate row 0..63 (i,f,g,o blocks of 16)
  const float* Wih = dir ? Wih_b : Wih_f;
  const float* Whh = dir ? Whh_b : Whh_f;
  const float* bih = dir ? bih_b : bih_f;
  const float* bhh = dir ? bhh_b : bhh_f;

  float wih[LH], whh[LH];
#pragma unroll
  for (int k = 0; k < LH; ++k) { wih[k] = Wih[r * LH + k]; whh[k] = Whh[r * LH + k]; }
  float bias = bih[r] + bhh[r];

  __shared__ float shx[LH], shh[LH], shg[4 * LH];
  float c = 0.0f, h = 0.0f;
  if (r < LH) shh[r] = 0.0f;

  for (int step = 0; step < TSTEPS; ++step) {
    int t = dir ? (TSTEPS - 1 - step) : step;
    if (r < LH) shx[r] = seq[((size_t)g * TSTEPS + t) * GOUT + r];
    __syncthreads();                       // shx + prev shh visible
    float acc = bias;
#pragma unroll
    for (int k = 0; k < LH; ++k) acc += wih[k] * shx[k] + whh[k] * shh[k];
    shg[r] = acc;
    __syncthreads();                       // all gates visible
    if (r < LH) {
      float ig = 1.0f / (1.0f + expf(-shg[r]));
      float fg = 1.0f / (1.0f + expf(-shg[LH + r]));
      float gg = tanhf(shg[2 * LH + r]);
      float og = 1.0f / (1.0f + expf(-shg[3 * LH + r]));
      c = fg * c + ig * gg;
      h = og * tanhf(c);
      shh[r] = h;
    }
  }
  if (r < LH) hn[(size_t)g * (2 * LH) + dir * LH + r] = h;
}

// ---------------- final MLP: out = relu(hn@Wm1 + bm1) @ Wm2 + bm2 ------------
__global__ __launch_bounds__(64)
void mlp_kernel(const float* __restrict__ hn, const float* __restrict__ Wm1,
                const float* __restrict__ bm1, const float* __restrict__ Wm2,
                const float* __restrict__ bm2, float* __restrict__ out) {
  int g = threadIdx.x;
  if (g >= NG) return;
  float v[2 * LH];
#pragma unroll
  for (int k = 0; k < 2 * LH; ++k) v[k] = hn[g * (2 * LH) + k];
  float acc = 0.0f;
  for (int j = 0; j < 128; ++j) {
    float a = bm1[j];
#pragma unroll
    for (int k = 0; k < 2 * LH; ++k) a += v[k] * Wm1[k * 128 + j];
    a = a > 0.0f ? a : 0.0f;
    acc += a * Wm2[j];
  }
  out[g] = acc + bm2[0];
}

// ---------------- host launcher ----------------
extern "C" void kernel_launch(void* const* d_in, const int* in_sizes, int n_in,
                              void* d_out, int out_size, void* d_ws, size_t ws_size,
                              hipStream_t stream) {
  (void)in_sizes; (void)n_in; (void)out_size; (void)ws_size;
  const float* x    = (const float*)d_in[0];
  const int*   ei   = (const int*)d_in[1];          // [2, E]
  const int*   srcE = ei;
  const int*   dstE = ei + EE;
  const float* W[5] = { (const float*)d_in[3], (const float*)d_in[5],
                        (const float*)d_in[7], (const float*)d_in[9],
                        (const float*)d_in[11] };
  const float* B[5] = { (const float*)d_in[4], (const float*)d_in[6],
                        (const float*)d_in[8], (const float*)d_in[10],
                        (const float*)d_in[12] };
  const float* Wih_f = (const float*)d_in[13];
  const float* Whh_f = (const float*)d_in[14];
  const float* bih_f = (const float*)d_in[15];
  const float* bhh_f = (const float*)d_in[16];
  const float* Wih_b = (const float*)d_in[17];
  const float* Whh_b = (const float*)d_in[18];
  const float* bih_b = (const float*)d_in[19];
  const float* bhh_b = (const float*)d_in[20];
  const float* Wm1   = (const float*)d_in[21];
  const float* bm1   = (const float*)d_in[22];
  const float* Wm2   = (const float*)d_in[23];
  const float* bm2   = (const float*)d_in[24];

  // carve workspace
  size_t off = 0;
  auto carve = [&](size_t bytes) {
    size_t o = off;
    off += (bytes + 255) & ~(size_t)255;
    return (char*)d_ws + o;
  };
  float*          bufA  = (float*)carve((size_t)NN * HID * 4);  // layer input / relu out
  float*          bufG  = (float*)carve((size_t)NN * HID * 4);  // GEMM output
  float*          bufAg = (float*)carve((size_t)NN * HID * 4);  // aggregation
  unsigned short* packP = (unsigned short*)carve((size_t)94 * 16 * 512 * 2);
  float*          deg   = (float*)carve((size_t)NN * 4);
  float*          invs  = (float*)carve((size_t)NN * 4);
  float*          invd  = (float*)carve((size_t)NN * 4);
  float*          hn    = (float*)carve((size_t)NG * 2 * LH * 4);

  // degrees
  zero_f32<<<CDIV(NN, 256), 256, 0, stream>>>(deg, (size_t)NN);
  deg_accum<<<CDIV(EE, 256), 256, 0, stream>>>(dstE, deg);
  deg_finalize<<<CDIV(NN, 256), 256, 0, stream>>>(deg, invs, invd);

  const int Kd[5] = { INDIM, HID, HID, HID, HID };
  const int Nd[5] = { HID, HID, HID, HID, GOUT };

  for (int l = 0; l < 5; ++l) {
    const int K = Kd[l], N = Nd[l];
    const int ktiles = CDIV(K, 32), ntiles = N / 16;
    const float* Ain = (l == 0) ? x : bufA;

    size_t packN = (size_t)ktiles * ntiles * 512;
    pack_w_bf16<<<CDIV(packN, 256), 256, 0, stream>>>(W[l], packP, K, N, ktiles, ntiles);

    if (N == HID) {
      // 2 M-tiles x 4 wave-columns x 4 N-tiles/wave: block covers M=32, N=256;
      // A streamed from HBM exactly once.
      gemm_wmma_bf16<2, 4, 4><<<dim3(NN / 32, 1), 256, 0, stream>>>(
          Ain, packP, bufG, K, N, ktiles, ntiles);
    } else {
      // 8 M-tiles x 1 N-tile
      gemm_wmma_bf16<8, 1, 1><<<dim3(NN / 128, 1), 256, 0, stream>>>(
          Ain, packP, bufG, K, N, ktiles, ntiles);
    }

    size_t nElems = (size_t)NN * N;
    zero_f32<<<CDIV(nElems, 256), 256, 0, stream>>>(bufAg, nElems);
    if (N == HID) {
      size_t sThreads = (size_t)EE * (HID / 4);
      edge_scatter<HID><<<CDIV(sThreads, 256), 256, 0, stream>>>(srcE, dstE, invs, bufG, bufAg);
      combine_relu<HID><<<CDIV(nElems, 256), 256, 0, stream>>>(bufAg, bufG, invd, B[l], bufA);
    } else {
      size_t sThreads = (size_t)EE * (GOUT / 4);
      edge_scatter<GOUT><<<CDIV(sThreads, 256), 256, 0, stream>>>(srcE, dstE, invs, bufG, bufAg);
      combine_relu<GOUT><<<CDIV(nElems, 256), 256, 0, stream>>>(bufAg, bufG, invd, B[l], bufA);
    }
  }

  // bufA now holds seq [NG, TSTEPS, GOUT]
  lstm_kernel<<<dim3(NG, 2), 64, 0, stream>>>(bufA, Wih_f, Whh_f, bih_f, bhh_f,
                                              Wih_b, Whh_b, bih_b, bhh_b, hn);
  mlp_kernel<<<1, 64, 0, stream>>>(hn, Wm1, bm1, Wm2, bm2, (float*)d_out);
}
